// SelfAttentionModule_65481071407281
// MI455X (gfx1250) — compile-verified, matrix operand order checked by reading, not run on hardware
//
#include <hip/hip_runtime.h>
#include <hip/hip_bf16.h>

typedef _Float16     v8h   __attribute__((ext_vector_type(8)));
typedef _Float16     v16h  __attribute__((ext_vector_type(16)));
typedef float        v8f   __attribute__((ext_vector_type(8)));
typedef unsigned int u32x4 __attribute__((ext_vector_type(4)));
typedef int          i32x4 __attribute__((ext_vector_type(4)));
typedef int          i32x8 __attribute__((ext_vector_type(8)));

#define WMMA_F16(a, b, c) \
  __builtin_amdgcn_wmma_f32_16x16x32_f16(false, (a), false, (b), (short)0, (c), false, false)

// ---------------------------------------------------------------------------
// Problem dims (fixed by the reference)
// ---------------------------------------------------------------------------
constexpr int BATCH = 4;
constexpr int SEQ   = 4096;
constexpr int EMB   = 1024;

// LDS B-tile geometry: 64 output-columns x 32 K, f16, padded row pitch.
// TDM pads 4 DWORDs (16B) after every 16 DWORDs (64B = one 32-elem row),
// giving an 80-byte (40 f16) pitch -> conflict-free ds_load_b128 fragments.
constexpr int TILE_N     = 64;
constexpr int TILE_K     = 32;
constexpr int LDS_PITCH  = 40;                    // f16 elements per row
constexpr int LDS_TILE   = TILE_N * LDS_PITCH;    // f16 elements per buffer

// ---------------------------------------------------------------------------
// Fragment loaders. CDNA5 16-bit A (16x32) / B (32x16) per-lane layout:
//   lanes 0..15  hold row/col (lane%16), K = {0..7} and {16..23}
//   lanes 16..31 hold row/col (lane%16), K = {8..15} and {24..31}
// Caller passes pointer already offset by  row*ld + 8*half.
// ---------------------------------------------------------------------------
__device__ __forceinline__ v16h ldfrag_h(const _Float16* p) {
  v8h lo = *(const v8h*)(p);
  v8h hi = *(const v8h*)(p + 16);
  v16h r;
#pragma unroll
  for (int i = 0; i < 8; ++i) { r[i] = lo[i]; r[i + 8] = hi[i]; }
  return r;
}

__device__ __forceinline__ v16h ldfrag_f(const float* p, float scale) {
  v16h r;
#pragma unroll
  for (int i = 0; i < 8; ++i) {
    r[i]     = (_Float16)(p[i]      * scale);
    r[i + 8] = (_Float16)(p[i + 16] * scale);
  }
  return r;
}

// ---------------------------------------------------------------------------
// Tensor Data Mover: stage a [TILE_N x TILE_K] f16 tile (row-major, leading
// dim ldb elements) from global into LDS with padding.  Issued by wave 0
// only; tracked with TENSORcnt.  D# built per CDNA5 ISA ch. 8.
// This toolchain exposes the 6-argument builtin:
//   (v4u group0, v8i group1, v4i group2, v4i group3, v8i extra, i32 cpol)
// ---------------------------------------------------------------------------
#if __has_builtin(__builtin_amdgcn_tensor_load_to_lds)
#define HAVE_TDM 1
__device__ __forceinline__ void tdm_issue(const _Float16* gsrc, unsigned lds_byte,
                                          unsigned ldb_elems) {
  const unsigned long long ga = (unsigned long long)(uintptr_t)gsrc;

  u32x4 g0;
  g0[0] = 1u;                                   // count=1, user descriptor
  g0[1] = lds_byte;                             // lds_addr
  g0[2] = (unsigned)(ga & 0xFFFFFFFFu);         // global_addr[31:0]
  g0[3] = (unsigned)((ga >> 32) & 0x1FFFFFFu)   // global_addr[56:32]
          | (2u << 30);                         // type = 2 ("image")

  i32x8 g1;
  g1[0] = (int)((1u << 16)                      // data_size = 1 (2 bytes)
                | (1u << 20)                    // pad_enable
                | (3u << 22)                    // pad_interval: 16 DWORDs
                | (3u << 25));                  // pad_amount:   4 DWORDs
  const unsigned td0 = 1u << 20;                // tensor_dim0 (no clipping)
  const unsigned td1 = 1u << 20;                // tensor_dim1 (no clipping)
  g1[1] = (int)((td0 & 0xFFFFu) << 16);         // [63:48] = tensor_dim0 lo16
  g1[2] = (int)((td0 >> 16) | ((td1 & 0xFFFFu) << 16));
  g1[3] = (int)((td1 >> 16) | ((unsigned)TILE_K << 16));   // tile_dim0 = 32
  g1[4] = TILE_N;                               // tile_dim1 = 64, tile_dim2 = 0
  g1[5] = (int)ldb_elems;                       // tensor_dim0_stride[31:0]
  g1[6] = 0;                                    // stride hi, dim1_stride lo
  g1[7] = 0;

  i32x4 z4 = {0, 0, 0, 0};
  i32x8 z8 = {0, 0, 0, 0, 0, 0, 0, 0};
  __builtin_amdgcn_tensor_load_to_lds(g0, g1, z4, z4, z8, 0);
}
#else
#define HAVE_TDM 0
#endif

// Fallback / portability: cooperative copy of the same tile with padding.
__device__ __forceinline__ void coop_stage(const _Float16* gsrc, _Float16* lbuf,
                                           unsigned ldb_elems, int tid) {
  // 64 rows x 32 f16 = 256 v8h chunks; 128 threads -> 2 chunks each.
#pragma unroll
  for (int c = tid; c < (TILE_N * TILE_K) / 8; c += 128) {
    const int row = c >> 2;            // 4 chunks per row
    const int off = (c & 3) * 8;
    *(v8h*)(lbuf + row * LDS_PITCH + off) =
        *(const v8h*)(gsrc + (size_t)row * ldb_elems + off);
  }
}

// ---------------------------------------------------------------------------
// Shared wave-level GEMM body:  C[128x64] per block (4 waves x 32 rows),
// A read straight from global (contiguous along K), B tile double-buffered
// in LDS via TDM.  acc[2][4] = 8 WMMAs per 32-wide K step.
// ---------------------------------------------------------------------------
__device__ __forceinline__ void gemm_body(const _Float16* __restrict__ A,
                                          const _Float16* __restrict__ B,
                                          int lda, int ldb, int kdim,
                                          int m0, int n0, _Float16* lds,
                                          int tid, v8f acc[2][4]) {
  const int wave = tid >> 5;
  const int lane = tid & 31;
  const int half = lane >> 4;
  const int r    = lane & 15;

  const _Float16* arow = A + (size_t)(m0 + r) * lda + 8 * half;
  const _Float16* btile = B + (size_t)n0 * ldb;
  const unsigned lds_base = (unsigned)(uintptr_t)lds;  // low 32 bits = LDS offset

  const int nsteps = kdim / TILE_K;

  // Prologue: stage tile 0 into buffer 0.
#if HAVE_TDM
  if (wave == 0) tdm_issue(btile, lds_base, (unsigned)ldb);
#else
  coop_stage(btile, lds, (unsigned)ldb, tid);
#endif

  for (int i = 0; i < nsteps; ++i) {
    const int k0 = i * TILE_K;
    // Kick off the next tile's DMA into the other buffer (freed by the
    // barrier at the bottom of iteration i-1).
#if HAVE_TDM
    if (wave == 0) {
      if (i + 1 < nsteps) {
        tdm_issue(btile + (size_t)(k0 + TILE_K),
                  lds_base + (unsigned)(((i + 1) & 1) * LDS_TILE * 2),
                  (unsigned)ldb);
        __builtin_amdgcn_s_wait_tensorcnt(1);  // oldest (current tile) done
      } else {
        __builtin_amdgcn_s_wait_tensorcnt(0);  // last tile done
      }
    }
#else
    if (i + 1 < nsteps)
      coop_stage(btile + (size_t)(k0 + TILE_K), lds + ((i + 1) & 1) * LDS_TILE,
                 (unsigned)ldb, tid);
#endif
    __syncthreads();  // publish: buffer (i&1) is ready

    __builtin_prefetch(arow + k0 + 8 * TILE_K, 0, 0);
    const v16h a0 = ldfrag_h(arow + k0);
    const v16h a1 = ldfrag_h(arow + 16 * lda + k0);

    const _Float16* buf = lds + (i & 1) * LDS_TILE;
#pragma unroll
    for (int t = 0; t < 4; ++t) {
      const v16h bf = ldfrag_h(buf + (t * 16 + r) * LDS_PITCH + 8 * half);
      acc[0][t] = WMMA_F16(a0, bf, acc[0][t]);
      acc[1][t] = WMMA_F16(a1, bf, acc[1][t]);
    }
    __syncthreads();  // all reads of buffer (i&1) done before it is reloaded
  }
}

// ---------------------------------------------------------------------------
// Kernel 1: QKV projection.  y = x @ W^T (torch Linear), f32 in, f16 out.
// grid = (EMB/64, (B*S)/64, 3), block = 128.  Q gets 1/sqrt(E) folded in.
// V is stored transposed per batch: Vt[b][e][s].
// ---------------------------------------------------------------------------
__global__ __launch_bounds__(128)
void proj_kernel(const float* __restrict__ x,
                 const float* __restrict__ Wq,
                 const float* __restrict__ Wk,
                 const float* __restrict__ Wv,
                 _Float16* __restrict__ Qh,
                 _Float16* __restrict__ Kh,
                 _Float16* __restrict__ Vt) {
  const int z = blockIdx.z;
  const float* W = (z == 0) ? Wq : (z == 1) ? Wk : Wv;
  const float a_scale = (z == 0) ? 0.03125f : 1.0f;  // 1/sqrt(1024)

  const int tid  = threadIdx.x;
  const int wave = tid >> 5;
  const int lane = tid & 31;
  const int half = lane >> 4;
  const int r    = lane & 15;

  const int m0 = blockIdx.y * 64 + wave * 16;
  const int n0 = blockIdx.x * 64;

  v8f acc[4] = {};
  const float* arow = x + (size_t)(m0 + r) * EMB + 8 * half;

  for (int k0 = 0; k0 < EMB; k0 += 32) {
    __builtin_prefetch(arow + k0 + 128, 0, 0);
    v16h afrag = ldfrag_f(arow + k0, a_scale);
#pragma unroll
    for (int t = 0; t < 4; ++t) {
      const float* brow = W + (size_t)(n0 + t * 16 + r) * EMB + k0 + 8 * half;
      acc[t] = WMMA_F16(afrag, ldfrag_f(brow, 1.0f), acc[t]);
    }
  }

#pragma unroll
  for (int t = 0; t < 4; ++t) {
#pragma unroll
    for (int rr = 0; rr < 8; ++rr) {
      const int m = m0 + rr + 8 * half;
      const int n = n0 + t * 16 + r;
      const _Float16 val = (_Float16)acc[t][rr];
      if (z == 0) {
        Qh[(size_t)m * EMB + n] = val;
      } else if (z == 1) {
        Kh[(size_t)m * EMB + n] = val;
      } else {
        const int b = m >> 12;
        const int s = m & (SEQ - 1);
        Vt[((size_t)b * EMB + n) * SEQ + s] = val;
      }
    }
  }
}

// ---------------------------------------------------------------------------
// Kernel 2: scores[b,q,k] = sum_e Qh[b,q,e]*Kh[b,k,e].
// grid = (SEQ/64, SEQ/128, BATCH), block = 128 (4 waves x 32 rows).
// ---------------------------------------------------------------------------
__global__ __launch_bounds__(128)
void scores_kernel(const _Float16* __restrict__ Qh,
                   const _Float16* __restrict__ Kh,
                   float* __restrict__ Sc) {
  __shared__ _Float16 lds[2 * LDS_TILE];

  const int b    = blockIdx.z;
  const int tid  = threadIdx.x;
  const int wave = tid >> 5;
  const int lane = tid & 31;
  const int half = lane >> 4;
  const int r    = lane & 15;

  const int m0 = blockIdx.y * 128 + wave * 32;
  const int n0 = blockIdx.x * 64;

  v8f acc[2][4] = {};
  gemm_body(Qh + (size_t)b * SEQ * EMB, Kh + (size_t)b * SEQ * EMB,
            EMB, EMB, EMB, m0, n0, lds, tid, acc);

  float* out = Sc + (size_t)b * SEQ * SEQ;
#pragma unroll
  for (int g = 0; g < 2; ++g)
#pragma unroll
    for (int t = 0; t < 4; ++t)
#pragma unroll
      for (int rr = 0; rr < 8; ++rr) {
        const int m = m0 + g * 16 + rr + 8 * half;
        const int n = n0 + t * 16 + r;
        out[(size_t)m * SEQ + n] = acc[g][t][rr];
      }
}

// ---------------------------------------------------------------------------
// Kernel 3: row softmax over 4096 f32 scores -> f16 probabilities.
// grid = (SEQ, BATCH), block = 256 (8 wave32s), 16 elements per thread.
// ---------------------------------------------------------------------------
__global__ __launch_bounds__(256)
void softmax_kernel(const float* __restrict__ Sc, _Float16* __restrict__ P) {
  const int row = blockIdx.x;
  const int b   = blockIdx.y;
  const float* srow = Sc + ((size_t)b * SEQ + row) * SEQ;
  _Float16*    prow = P  + ((size_t)b * SEQ + row) * SEQ;

  const int tid  = threadIdx.x;
  const int wave = tid >> 5;
  const int lane = tid & 31;

  __shared__ float red_max[8];
  __shared__ float red_sum[8];

  float v[16];
  float m = -3.402823466e38f;
#pragma unroll
  for (int i = 0; i < 16; ++i) {
    v[i] = srow[tid + i * 256];
    m = fmaxf(m, v[i]);
  }
#pragma unroll
  for (int off = 16; off > 0; off >>= 1) m = fmaxf(m, __shfl_xor(m, off, 32));
  if (lane == 0) red_max[wave] = m;
  __syncthreads();
  if (wave == 0) {
    float t = (lane < 8) ? red_max[lane] : -3.402823466e38f;
#pragma unroll
    for (int off = 4; off > 0; off >>= 1) t = fmaxf(t, __shfl_xor(t, off, 32));
    if (lane == 0) red_max[0] = t;
  }
  __syncthreads();
  m = red_max[0];

  float s = 0.0f;
#pragma unroll
  for (int i = 0; i < 16; ++i) {
    v[i] = __expf(v[i] - m);
    s += v[i];
  }
#pragma unroll
  for (int off = 16; off > 0; off >>= 1) s += __shfl_xor(s, off, 32);
  if (lane == 0) red_sum[wave] = s;
  __syncthreads();
  if (wave == 0) {
    float t = (lane < 8) ? red_sum[lane] : 0.0f;
#pragma unroll
    for (int off = 4; off > 0; off >>= 1) t += __shfl_xor(t, off, 32);
    if (lane == 0) red_sum[0] = t;
  }
  __syncthreads();
  const float inv = 1.0f / red_sum[0];

#pragma unroll
  for (int i = 0; i < 16; ++i) prow[tid + i * 256] = (_Float16)(v[i] * inv);
}

// ---------------------------------------------------------------------------
// Kernel 4: out[b,q,e] = sum_k P[b,q,k]*V[b,k,e] using Vt[b][e][k].
// grid = (EMB/64, SEQ/128, BATCH), block = 128.
// ---------------------------------------------------------------------------
__global__ __launch_bounds__(128)
void out_kernel(const _Float16* __restrict__ P,
                const _Float16* __restrict__ Vt,
                float* __restrict__ O) {
  __shared__ _Float16 lds[2 * LDS_TILE];

  const int b    = blockIdx.z;
  const int tid  = threadIdx.x;
  const int wave = tid >> 5;
  const int lane = tid & 31;
  const int half = lane >> 4;
  const int r    = lane & 15;

  const int m0 = blockIdx.y * 128 + wave * 32;
  const int n0 = blockIdx.x * 64;

  v8f acc[2][4] = {};
  gemm_body(P + (size_t)b * SEQ * SEQ, Vt + (size_t)b * EMB * SEQ,
            SEQ, SEQ, SEQ, m0, n0, lds, tid, acc);

#pragma unroll
  for (int g = 0; g < 2; ++g)
#pragma unroll
    for (int t = 0; t < 4; ++t)
#pragma unroll
      for (int rr = 0; rr < 8; ++rr) {
        const int m = m0 + g * 16 + rr + 8 * half;
        const int n = n0 + t * 16 + r;
        O[((size_t)b * SEQ + m) * EMB + n] = acc[g][t][rr];
      }
}

// ---------------------------------------------------------------------------
// Host-side launch
// ---------------------------------------------------------------------------
extern "C" void kernel_launch(void* const* d_in, const int* in_sizes, int n_in,
                              void* d_out, int out_size, void* d_ws, size_t ws_size,
                              hipStream_t stream) {
  const float* x  = (const float*)d_in[0];
  const float* Wq = (const float*)d_in[1];
  const float* Wk = (const float*)d_in[2];
  const float* Wv = (const float*)d_in[3];
  float* out = (float*)d_out;

  constexpr size_t QKV_BYTES = (size_t)BATCH * SEQ * EMB * sizeof(_Float16);  // 32 MB
  constexpr size_t SC_BYTES  = (size_t)BATCH * SEQ * SEQ * sizeof(float);     // 256 MB
  char* ws = (char*)d_ws;
  _Float16* Qh = (_Float16*)(ws);
  _Float16* Kh = (_Float16*)(ws + QKV_BYTES);
  _Float16* Vt = (_Float16*)(ws + 2 * QKV_BYTES);
  float*    Sc = (float*)   (ws + 3 * QKV_BYTES);
  _Float16* Pp = (_Float16*)(ws + 3 * QKV_BYTES + SC_BYTES);

  {
    dim3 grid(EMB / 64, (BATCH * SEQ) / 64, 3);
    proj_kernel<<<grid, 128, 0, stream>>>(x, Wq, Wk, Wv, Qh, Kh, Vt);
  }
  {
    dim3 grid(SEQ / 64, SEQ / 128, BATCH);
    scores_kernel<<<grid, 128, 0, stream>>>(Qh, Kh, Sc);
  }
  {
    dim3 grid(SEQ, BATCH, 1);
    softmax_kernel<<<grid, 256, 0, stream>>>(Sc, Pp);
  }
  {
    dim3 grid(EMB / 64, SEQ / 128, BATCH);
    out_kernel<<<grid, 128, 0, stream>>>(Pp, Vt, out);
  }
}
